// CCGBlock_10179072492002
// MI455X (gfx1250) — matrix operand order repaired and checked by reading, not compiled, not run on hardware
//
#include <hip/hip_runtime.h>
#include <hip/hip_bf16.h>

// CCGBlock fused pipeline for MI455X (gfx1250), wave32 + WMMA f32_16x16x32_f16.
// B=64, N=512, F_IN=F_OUT=512, E=4, GSD=1024.

#define EPS_ 1e-5f

typedef __attribute__((ext_vector_type(16))) _Float16 v16h;
typedef __attribute__((ext_vector_type(8)))  _Float16 v8h;
typedef __attribute__((ext_vector_type(8)))  float    v8f;
typedef __attribute__((ext_vector_type(4)))  float    v4f;

#define WMMA(a,b,c) __builtin_amdgcn_wmma_f32_16x16x32_f16(false,(a),false,(b),(short)0,(c),false,false)

// A operand (16x32 f16, M x K): lane holds row m = lane&15;
// v16h elems 0..7  = A[m][k + 8*hi + 0..7]
// v16h elems 8..15 = A[m][k + 16 + 8*hi + 0..7]   (hi = lane>>4)
// Source is f32 row-major; convert on load.
__device__ __forceinline__ v16h load_a_f32(const float* __restrict__ row, int k, int hi) {
  const float* p0 = row + k + 8*hi;
  const float* p1 = row + k + 16 + 8*hi;
  v4f x0 = *(const v4f*)(p0);
  v4f x1 = *(const v4f*)(p0+4);
  v4f y0 = *(const v4f*)(p1);
  v4f y1 = *(const v4f*)(p1+4);
  v16h a;
#pragma unroll
  for (int i=0;i<4;++i){
    a[i]    = (_Float16)x0[i];
    a[4+i]  = (_Float16)x1[i];
    a[8+i]  = (_Float16)y0[i];
    a[12+i] = (_Float16)y1[i];
  }
  return a;
}

// B operand (32x16 f16, K x N): lane holds column n = lane&15;
// v16h elems 0..7 = B[k + 8*hi + 0..7][n], elems 8..15 = B[k + 16 + 8*hi + 0..7][n].  With B stored transposed (BT[n][k],
// K-contiguous f16) this is two b128 loads.
__device__ __forceinline__ v16h load_b_f16(const _Float16* __restrict__ col, int k, int hi) {
  const _Float16* p = col + k + 8*hi;
  v8h b0 = *(const v8h*)(p);
  v8h b1 = *(const v8h*)(p+16);
  v16h b;
#pragma unroll
  for (int i=0;i<8;++i){ b[i]=b0[i]; b[8+i]=b1[i]; }
  return b;
}

// ---------- kernel 1: transpose+convert weights to f16 (tiny, once) ----------
__global__ void k_prep(const float* __restrict__ Wgfc, const float* __restrict__ Wgc,
                       _Float16* __restrict__ WgfcT, _Float16* __restrict__ WT1,
                       _Float16* __restrict__ WT2) {
  long i = (long)blockIdx.x*blockDim.x + threadIdx.x;
  const long NG = 512L*512;
  const long NW = 4L*512*512;
  if (i < NG) {
    long o = i >> 9, g = i & 511;
    WgfcT[i] = (_Float16)Wgfc[g*512 + o];                 // WgfcT[o][g]
  } else if (i < NG + NW) {
    long j = i - NG;
    long e = j / (512*512), rr = j % (512*512), o = rr >> 9, f = rr & 511;
    WT1[j] = (_Float16)Wgc[(e*1024 + f)*512 + o];         // WT1[e][o][f], f<512
  } else if (i < NG + 2*NW) {
    long j = i - NG - NW;
    long e = j / (512*512), rr = j % (512*512), o = rr >> 9, f = rr & 511;
    WT2[j] = (_Float16)Wgc[(e*1024 + 512 + f)*512 + o];   // WT2[e][o][f], rows 512..1023
  }
}

// ---------- kernel 2: gt = gs @ W_gfc + b_gfc   (128 x 512 x 512) ----------
__global__ void k_gt(const float* __restrict__ gs, const _Float16* __restrict__ WgfcT,
                     const float* __restrict__ bgfc, float* __restrict__ gt) {
  int wid  = (blockIdx.x * blockDim.x + threadIdx.x) >> 5;   // 64 waves
  int lane = threadIdx.x & 31, ln = lane & 15, hi = lane >> 4;
  int mt = wid >> 3, ot = wid & 7;
  int m0 = mt*16, o0 = ot*64;
  const float* arow = gs + (long)(m0 + ln)*512;
  v8f acc[4] = {};
  for (int k=0;k<512;k+=32) {
    v16h a = load_a_f32(arow, k, hi);
#pragma unroll
    for (int t=0;t<4;++t) {
      v16h b = load_b_f16(WgfcT + (long)(o0 + 16*t + ln)*512, k, hi);
      acc[t] = WMMA(a,b,acc[t]);
    }
  }
#pragma unroll
  for (int t=0;t<4;++t) {
    int o = o0 + 16*t + ln;
    float bias = bgfc[o];
#pragma unroll
    for (int v=0; v<8; ++v) {
      int m = m0 + v + 8*hi;
      gt[(long)m*512 + o] = acc[t][v] + bias;
    }
  }
}

// ---------- kernel 3: g_out = relu(BN(gt))  (stats over 128 rows) ----------
__global__ void k_gout(const float* __restrict__ gt, const float* __restrict__ gamma_g,
                       const float* __restrict__ beta_g, float* __restrict__ outg) {
  int o = blockIdx.x * blockDim.x + threadIdx.x;
  if (o >= 512) return;
  float s=0.f, q=0.f;
  for (int r=0;r<128;++r){ float v = gt[(long)r*512+o]; s+=v; q+=v*v; }
  float mean = s * (1.f/128.f);
  float var  = q * (1.f/128.f) - mean*mean;
  float sc = rsqrtf(var + EPS_) * gamma_g[o];
  float bt = beta_g[o];
  for (int r=0;r<128;++r){
    float v = (gt[(long)r*512+o]-mean)*sc + bt;
    outg[(long)r*512+o] = fmaxf(v, 0.f);
  }
}

// ---------- kernel 4: c[(m*4+e)*512+o] = gt @ W2_e   (per e: 128x512x512) ----------
__global__ void k_corr(const float* __restrict__ gt, const _Float16* __restrict__ WT2,
                       float* __restrict__ cbuf) {
  int wid  = (blockIdx.x*blockDim.x + threadIdx.x) >> 5;     // 256 waves
  int lane = threadIdx.x & 31, ln = lane&15, hi = lane>>4;
  int e = wid >> 6, rem = wid & 63, mt = rem >> 3, ot = rem & 7;
  int m0 = mt*16, o0 = ot*64;
  const float* arow = gt + (long)(m0+ln)*512;
  const _Float16* bt = WT2 + (long)e*512*512;
  v8f acc[4] = {};
  for (int k=0;k<512;k+=32) {
    v16h a = load_a_f32(arow, k, hi);
#pragma unroll
    for (int t=0;t<4;++t) {
      v16h b = load_b_f16(bt + (long)(o0 + 16*t + ln)*512, k, hi);
      acc[t] = WMMA(a,b,acc[t]);
    }
  }
#pragma unroll
  for (int t=0;t<4;++t) {
    int o = o0 + 16*t + ln;
#pragma unroll
    for (int v=0; v<8; ++v) {
      int m = m0 + v + 8*hi;
      cbuf[((long)m*4 + e)*512 + o] = acc[t][v];
    }
  }
}

// ---------- kernel 5: supportT[b,e,o,m] = f16( x[b] @ W1_e + sel(m) ) ----------
__global__ void k_support(const float* __restrict__ x, const _Float16* __restrict__ WT1,
                          const float* __restrict__ cbuf, const long long* __restrict__ sub,
                          _Float16* __restrict__ supT) {
  int wid  = (blockIdx.x * blockDim.x + threadIdx.x) >> 5;   // 65536 waves
  int lane = threadIdx.x & 31, ln = lane & 15, hi = lane >> 4;
  int b  = wid >> 10;
  int e  = (wid >> 8) & 3;
  int mt = (wid >> 3) & 31;
  int ot = wid & 7;
  int m0 = mt*16, o0 = ot*64;
  const float* arow = x + ((long)b*512 + m0 + ln)*512;
  const _Float16* bt = WT1 + (long)e*512*512;
  v8f acc[4] = {};
  for (int k=0;k<512;k+=32) {
    v16h a = load_a_f32(arow, k, hi);
#pragma unroll
    for (int t=0;t<4;++t) {
      v16h bb = load_b_f16(bt + (long)(o0 + 16*t + ln)*512, k, hi);
      acc[t] = WMMA(a,bb,acc[t]);
    }
  }
  int n1  = (int)sub[2*b];
  int n12 = n1 + (int)sub[2*b+1];
  const float* c0 = cbuf + ((long)(b*2)*4 + e)*512;
  const float* c1 = c0 + 4*512;
#pragma unroll
  for (int t=0;t<4;++t) {
    int o = o0 + 16*t + ln;
    float c0v = c0[o], c1v = c1[o];
    v8h h;
#pragma unroll
    for (int v=0; v<8; ++v) {
      int m = m0 + 8*hi + v;                         // C elem v -> row m0 + v + 8*hi
      float add = (m < n1) ? c0v : ((m < n12) ? c1v : 0.f);
      h[v] = (_Float16)(acc[t][v] + add);
    }
    // transposed f16 store: supportT[(b*4+e)][o][m], 8 consecutive m -> one b128
    *(v8h*)(supT + (((long)(b*4+e)*512 + o)*512 + m0 + 8*hi)) = h;
  }
}

// ---------- kernel 6: x_out_raw = sum_e adj[b,e] @ support[b,e] + b_gc ----------
__global__ void k_xout(const float* __restrict__ adj, const _Float16* __restrict__ supT,
                       const float* __restrict__ bgc, float* __restrict__ out) {
  int b  = blockIdx.x >> 4;          // 16 blocks per batch (4x4 tiles of 128x128)
  int r  = blockIdx.x & 15;
  int mt = r >> 2, ot = r & 3;
  int w  = threadIdx.x >> 5;
  int lane = threadIdx.x & 31, ln = lane & 15, hi = lane >> 4;
  int m0 = mt*128 + w*16;
  int o0 = ot*128;
  v8f acc[8] = {};
  for (int e=0;e<4;++e) {
    const float* arow = adj + (((long)(b*4+e)*512) + m0 + ln)*512;
    const _Float16* bt = supT + ((long)(b*4+e)*512)*512;
    for (int k=0;k<512;k+=32) {
      __builtin_prefetch(arow + k + 64, 0, 1);       // global_prefetch_b8
      v16h a = load_a_f32(arow, k, hi);
#pragma unroll
      for (int t=0;t<8;++t) {
        v16h bb = load_b_f16(bt + (long)(o0 + 16*t + ln)*512, k, hi);
        acc[t] = WMMA(a,bb,acc[t]);
      }
    }
  }
#pragma unroll
  for (int t=0;t<8;++t) {
    int o = o0 + 16*t + ln;
    float bias = bgc[o];
#pragma unroll
    for (int v=0; v<8; ++v) {
      int m = m0 + v + 8*hi;
      out[((long)b*512 + m)*512 + o] = acc[t][v] + bias;
    }
  }
}

// ---------- kernel 7: deterministic per-feature mean/var over 32768 rows ----------
__global__ void k_stats(const float* __restrict__ out, float* __restrict__ stats) {
  __shared__ float ssum[256], ssq[256];
  int o = blockIdx.x, tid = threadIdx.x;
  float s=0.f, q=0.f;
  for (long r=tid; r<32768; r+=256){ float v = out[r*512+o]; s+=v; q+=v*v; }
  ssum[tid]=s; ssq[tid]=q; __syncthreads();
  for (int st=128; st>0; st>>=1){
    if (tid<st){ ssum[tid]+=ssum[tid+st]; ssq[tid]+=ssq[tid+st]; }
    __syncthreads();
  }
  if (tid==0){
    float m = ssum[0]*(1.f/32768.f);
    stats[o]     = m;
    stats[512+o] = ssq[0]*(1.f/32768.f) - m*m;
  }
}

// ---------- kernel 8: x_out = relu(BN(raw) * mask) ----------
__global__ void k_norm(float* __restrict__ out, const float* __restrict__ stats,
                       const float* __restrict__ gamma_n, const float* __restrict__ beta_n,
                       const float* __restrict__ mask) {
  long i = (long)blockIdx.x*blockDim.x + threadIdx.x;
  if (i >= 16777216L) return;
  int  o  = (int)(i & 511);
  long bn = i >> 9;                                  // b*512 + n
  float raw  = out[i];
  float v = (raw - stats[o]) * rsqrtf(stats[512+o] + EPS_) * gamma_n[o] + beta_n[o];
  v *= mask[bn];
  out[i] = fmaxf(v, 0.f);
}

extern "C" void kernel_launch(void* const* d_in, const int* in_sizes, int n_in,
                              void* d_out, int out_size, void* d_ws, size_t ws_size,
                              hipStream_t stream) {
  (void)in_sizes; (void)n_in; (void)out_size; (void)ws_size;
  const float*     x     = (const float*)d_in[0];
  const float*     adj   = (const float*)d_in[1];
  const float*     gs    = (const float*)d_in[2];
  const long long* sub   = (const long long*)d_in[3];   // int64
  const float*     mask  = (const float*)d_in[4];
  const float*     Wgfc  = (const float*)d_in[5];
  const float*     bgfc  = (const float*)d_in[6];
  const float*     Wgc   = (const float*)d_in[7];
  const float*     bgc   = (const float*)d_in[8];
  const float*     gamn  = (const float*)d_in[9];
  const float*     betn  = (const float*)d_in[10];
  const float*     gamg  = (const float*)d_in[11];
  const float*     betg  = (const float*)d_in[12];
  float* out = (float*)d_out;

  char* w = (char*)d_ws;
  auto al = [](size_t v){ return (v + 255) & ~(size_t)255; };
  size_t off = 0;
  _Float16* WgfcT = (_Float16*)(w + off); off = al(off + 512L*512*2);
  _Float16* WT1   = (_Float16*)(w + off); off = al(off + 4L*512*512*2);
  _Float16* WT2   = (_Float16*)(w + off); off = al(off + 4L*512*512*2);
  float*    gt    = (float*)  (w + off);  off = al(off + 128L*512*4);
  float*    cbuf  = (float*)  (w + off);  off = al(off + 128L*4*512*4);
  float*    stats = (float*)  (w + off);  off = al(off + 1024L*4);
  _Float16* supT  = (_Float16*)(w + off); // 64*4*512*512 f16 = 128 MB

  k_prep   <<<9216, 256, 0, stream>>>(Wgfc, Wgc, WgfcT, WT1, WT2);
  k_gt     <<<   8, 256, 0, stream>>>(gs, WgfcT, bgfc, gt);
  k_gout   <<<   2, 256, 0, stream>>>(gt, gamg, betg, out + 16777216L);
  k_corr   <<<  32, 256, 0, stream>>>(gt, WT2, cbuf);
  k_support<<<8192, 256, 0, stream>>>(x, WT1, cbuf, sub, supT);
  k_xout   <<<1024, 256, 0, stream>>>(adj, supT, bgc, out);
  k_stats  <<< 512, 256, 0, stream>>>(out, stats);
  k_norm   <<<65536,256, 0, stream>>>(out, stats, gamn, betn, mask);
}
